// Attention_43696997269586
// MI455X (gfx1250) — matrix operand order repaired and checked by reading, compiled, not run on hardware
//
#include <hip/hip_runtime.h>
#include <hip/hip_bf16.h>
#include <stdint.h>

// ---------------------------------------------------------------------------
// Attention block for MI455X (gfx1250): all matmuls on v_wmma_f32_16x16x32_bf16,
// tile staging via Tensor Data Mover (tensor_load_to_lds, TENSORcnt).
// B=4, T=2048, C=1024, H=16, D=64
// ---------------------------------------------------------------------------

typedef __attribute__((ext_vector_type(16))) __bf16        v16bf;
typedef __attribute__((ext_vector_type(8)))  float         v8f;
typedef __attribute__((ext_vector_type(4)))  unsigned int  u32x4;
typedef __attribute__((ext_vector_type(2)))  unsigned int  u32x2;
typedef __attribute__((ext_vector_type(8)))  int           i32x8;
typedef __attribute__((ext_vector_type(4)))  int           i32x4;

#define B_  4
#define T_  2048
#define C_  1024
#define H_  16
#define D_  64
#define BT_ (B_ * T_)

__device__ __forceinline__ unsigned short f2bf(float f) {
  unsigned int u = __float_as_uint(f);
  u += 0x7FFFu + ((u >> 16) & 1u);       // round-to-nearest-even
  return (unsigned short)(u >> 16);
}

union FragU { u32x4 q[2]; v16bf v; };
union PackU { u32x4 q; unsigned short s[8]; };

// A-matrix fragment (16-bit, 16xK): lanes 0-15 -> dwords {0..3, 8..11},
// lanes 16-31 -> dwords {4..7, 12..15} (ISA 7.12.2). rowBase must be 16B aligned.
__device__ __forceinline__ v16bf load_frag_a(const unsigned short* rowBase, int dwOff, int lane) {
  const unsigned int* p = (const unsigned int*)rowBase + dwOff + ((lane & 16) ? 4 : 0);
  FragU f;
  f.q[0] = *(const u32x4*)(p);
  f.q[1] = *(const u32x4*)(p + 8);
  return f.v;
}

// B-matrix fragment (16-bit, Kx16 column): lanes 0-15 -> dwords {0..3, 8..11},
// lanes 16-31 -> dwords {4..7, 12..15} (same K order as the A fragment).
__device__ __forceinline__ v16bf load_frag_b(const unsigned short* rowBase, int dwOff, int lane) {
  const unsigned int* p = (const unsigned int*)rowBase + dwOff + ((lane & 16) ? 4 : 0);
  FragU f;
  f.q[0] = *(const u32x4*)(p);
  f.q[1] = *(const u32x4*)(p + 8);
  return f.v;
}

#define ZERO8 (v8f){0.f, 0.f, 0.f, 0.f, 0.f, 0.f, 0.f, 0.f}

// 16-lane butterfly reductions on the VALU via DPP ROW_XMASK (0x160 | mask).
template <int M>
__device__ __forceinline__ float xmask(float x) {
  return __int_as_float(
      __builtin_amdgcn_update_dpp(0, __float_as_int(x), 0x160 | M, 0xf, 0xf, false));
}
__device__ __forceinline__ float red16_max(float x) {
  x = fmaxf(x, xmask<1>(x));
  x = fmaxf(x, xmask<2>(x));
  x = fmaxf(x, xmask<4>(x));
  x = fmaxf(x, xmask<8>(x));
  return x;
}
__device__ __forceinline__ float red16_sum(float x) {
  x += xmask<1>(x);
  x += xmask<2>(x);
  x += xmask<4>(x);
  x += xmask<8>(x);
  return x;
}

// ---------------------------------------------------------------------------
// Tensor Data Mover: async 2D tile load global -> LDS (ISA ch.8 D# layout).
// tile_d0 elements contiguous per row, tile_d1 rows, rowStride in elements,
// 2-byte elements. LDS padding: pad PA+1 dwords after every 2<<PI dwords.
// ---------------------------------------------------------------------------
__device__ __forceinline__ void tdm_load_2d(unsigned ldsAddr, const void* gaddr,
                                            int tile_d0, int tile_d1,
                                            unsigned rowStride,
                                            int padInterval, int padAmount) {
  unsigned long long ga = (unsigned long long)(uintptr_t)gaddr;
  u32x4 g0;
  g0[0] = 1u;                                    // count=1, user-mode load
  g0[1] = ldsAddr;                               // D#.lds_addr
  g0[2] = (unsigned)ga;                          // global_addr[31:0]
  g0[3] = (unsigned)((ga >> 32) & 0x01FFFFFFu)   // global_addr[56:32]
          | 0x80000000u;                         // type=2 ("image")
  i32x8 g1;
  g1[0] = (int)((1u << 16)                       // data_size = 2 bytes
                | (1u << 20)                     // pad_enable
                | ((unsigned)padInterval << 22)
                | ((unsigned)padAmount << 25));
  g1[1] = 0;                                     // atomic_barrier=0; tensor_dim0 lo16=0
  g1[2] = (int)(0x4000u);                        // tensor_dim0 = 0x40000000 (no OOB)
  g1[3] = (int)(0x4000u | ((unsigned)tile_d0 << 16));   // tensor_dim1 hi; tile_dim0
  g1[4] = (int)((unsigned)tile_d1 & 0xFFFFu);    // tile_dim1; tile_dim2=0
  g1[5] = (int)rowStride;                        // tensor_dim0_stride lo32
  g1[6] = 0;                                     // stride hi16; dim1_stride lo16
  g1[7] = 0;
  i32x4 gz4 = (i32x4){0, 0, 0, 0};
  i32x8 gz8 = (i32x8){0, 0, 0, 0, 0, 0, 0, 0};
  __builtin_amdgcn_tensor_load_to_lds(g0, g1, gz4, gz4, gz8, 0);
}

__device__ __forceinline__ unsigned ldsof(const void* p) {
  return (unsigned)(uintptr_t)p;   // flat LDS address truncates to LDS byte offset
}

// ---------------------------------------------------------------------------
// f32 -> bf16 conversion
// ---------------------------------------------------------------------------
__global__ __launch_bounds__(256) void attn_cvt_bf16(const float* __restrict__ src,
                                                     unsigned short* __restrict__ dst, int n) {
  int i = blockIdx.x * blockDim.x + threadIdx.x;
  int stride = gridDim.x * blockDim.x;
  for (; i < n; i += stride) dst[i] = f2bf(src[i]);
}

// ---------------------------------------------------------------------------
// GEMM: Y = A(bf16 [M,K]) @ W(bf16 [N,K])^T , M=8192, N=K=1024.
// MODE 0: out bf16 in head-split [B,H,T,D] layout, scaled.
// MODE 1: out f32 row-major [M,N] with bias (final projection).
// Block tile 128x128, 8 waves of 64x32; TDM-fed double-buffered LDS.
// ---------------------------------------------------------------------------
#define GS 40   // LDS row stride in ushorts = 64B data + 16B TDM pad (PI=3, PA=3)

template <int MODE>
__global__ __launch_bounds__(256)
void attn_gemm_wmma(const unsigned short* __restrict__ A,
                    const unsigned short* __restrict__ W,
                    void* __restrict__ out,
                    const float* __restrict__ bias,
                    float scale) {
  const int tid  = threadIdx.x;
  const int lane = tid & 31;
  const int wv   = tid >> 5;
  const int m0   = (wv & 1) * 64;
  const int n0   = (wv >> 1) * 32;
  const int blockM = blockIdx.y * 128;
  const int blockN = blockIdx.x * 128;

  __shared__ __align__(16) unsigned short As[2][128 * GS];
  __shared__ __align__(16) unsigned short Bs[2][128 * GS];

  const unsigned short* gA = A + (size_t)blockM * C_;
  const unsigned short* gB = W + (size_t)blockN * C_;

  v8f acc[4][2];
#pragma unroll
  for (int mi = 0; mi < 4; ++mi)
#pragma unroll
    for (int ni = 0; ni < 2; ++ni) acc[mi][ni] = ZERO8;

  if (wv == 0) {   // kick off tile 0 DMA
    tdm_load_2d(ldsof(As[0]), gA, 32, 128, C_, 3, 3);
    tdm_load_2d(ldsof(Bs[0]), gB, 32, 128, C_, 3, 3);
  }

  int p = 0;
  for (int k0 = 0; k0 < C_; k0 += 32, p ^= 1) {
    if (wv == 0) __builtin_amdgcn_s_wait_tensorcnt(0);   // buffer p landed
    __syncthreads();   // all waves: buffer p ready; buffer p^1 no longer read
    if (wv == 0 && k0 + 32 < C_) {   // DMA next tile while computing this one
      tdm_load_2d(ldsof(As[p ^ 1]), gA + (k0 + 32), 32, 128, C_, 3, 3);
      tdm_load_2d(ldsof(Bs[p ^ 1]), gB + (k0 + 32), 32, 128, C_, 3, 3);
    }

    v16bf bfrag[2];
#pragma unroll
    for (int ni = 0; ni < 2; ++ni)
      bfrag[ni] = load_frag_b(Bs[p] + (n0 + ni * 16 + (lane & 15)) * GS, 0, lane);
#pragma unroll
    for (int mi = 0; mi < 4; ++mi) {
      v16bf afrag = load_frag_a(As[p] + (m0 + mi * 16 + (lane & 15)) * GS, 0, lane);
#pragma unroll
      for (int ni = 0; ni < 2; ++ni)
        acc[mi][ni] = __builtin_amdgcn_wmma_f32_16x16x32_bf16(
            false, afrag, false, bfrag[ni], (short)0, acc[mi][ni], false, false);
    }
  }

  const int rhalf = (lane >> 4) * 8;
#pragma unroll
  for (int mi = 0; mi < 4; ++mi) {
#pragma unroll
    for (int ni = 0; ni < 2; ++ni) {
      const int gn = blockN + n0 + ni * 16 + (lane & 15);
#pragma unroll
      for (int r = 0; r < 8; ++r) {
        const int gm = blockM + m0 + mi * 16 + rhalf + r;
        float val = acc[mi][ni][r] * scale;
        if (MODE == 0) {
          int b = gm >> 11, t = gm & (T_ - 1);
          int h = gn >> 6,  d = gn & 63;
          ((unsigned short*)out)[(((size_t)b * H_ + h) * T_ + t) * D_ + d] = f2bf(val);
        } else {
          ((float*)out)[(size_t)gm * C_ + gn] = val + bias[gn];
        }
      }
    }
  }
}

// ---------------------------------------------------------------------------
// Flash attention: Q,K,V bf16 [B*H, T, D]; O bf16 [B*T, C].
// Block = 128 q rows of one (b,h); 8 waves x 16 q rows; 64-key tiles (32 iters).
// K tiles via TDM; V transposed manually (TDM has no transpose); P staged
// per-wave with packed B64 stores using the key permutation q=(key&15)*4+key>>4.
// ---------------------------------------------------------------------------
#define FS 72   // LDS row stride in ushorts = 128B data + 16B pad (PI=4, PA=3)

__global__ __launch_bounds__(256)
void attn_flash_wmma(const unsigned short* __restrict__ Q,
                     const unsigned short* __restrict__ K,
                     const unsigned short* __restrict__ V,
                     unsigned short* __restrict__ O) {
  const int tid  = threadIdx.x;
  const int lane = tid & 31;
  const int wv   = tid >> 5;
  const int bh   = blockIdx.y;
  const int qw   = blockIdx.x * 128 + wv * 16;   // wave's first query row

  __shared__ __align__(16) unsigned short Kt[2][64 * FS];   // [key][d] via TDM
  __shared__ __align__(16) unsigned short Vt[2][64 * FS];   // [d][keypos q]
  __shared__ __align__(16) unsigned short Ps[8][16 * FS];   // per-wave P tile

  const size_t headBase = (size_t)bh * T_ * D_;
  const unsigned short* gKbase = K + headBase;

  v16bf qfrag[2];
  {
    const unsigned short* qrow = Q + headBase + (size_t)(qw + (lane & 15)) * D_;
    qfrag[0] = load_frag_a(qrow, 0, lane);
    qfrag[1] = load_frag_a(qrow, 16, lane);
  }

  // V staging: two 16B chunks per thread (64 keys x 64 d tile), transposed store
  const int key0  = tid >> 3;          // 0..31 ; second chunk at key0+32
  const int dbase = (tid & 7) * 8;
  const int q0    = ((key0 & 15) << 2) | (key0 >> 4);   // q(key0); q(key0+32)=q0+2
  const unsigned short* gV = V + headBase + (size_t)key0 * D_ + dbase;
  u32x4 vr0 = *(const u32x4*)gV, vr1 = *(const u32x4*)(gV + 32 * D_);

  if (wv == 0)   // kick off K tile 0 DMA
    tdm_load_2d(ldsof(Kt[0]), gKbase, D_, 64, D_, 4, 3);

  float mst[8], lst[8];
  v8f oacc[4];
#pragma unroll
  for (int r = 0; r < 8; ++r) { mst[r] = -1e30f; lst[r] = 0.f; }
#pragma unroll
  for (int nt = 0; nt < 4; ++nt) oacc[nt] = ZERO8;

  unsigned short* pbuf = Ps[wv];
  int p = 0;
  for (int kb = 0; kb < T_; kb += 64, p ^= 1) {
    // stage V regs -> Vt[p] (transposed, q-order)
    {
      PackU a, b2; a.q = vr0; b2.q = vr1;
#pragma unroll
      for (int j = 0; j < 8; ++j) {
        Vt[p][(dbase + j) * FS + q0]     = a.s[j];
        Vt[p][(dbase + j) * FS + q0 + 2] = b2.s[j];
      }
    }
    if (kb + 64 < T_) {   // issue next V tile's global loads
      gV += 64 * D_;
      vr0 = *(const u32x4*)gV; vr1 = *(const u32x4*)(gV + 32 * D_);
    }
    if (wv == 0) __builtin_amdgcn_s_wait_tensorcnt(0);   // Kt[p] landed
    __syncthreads();   // Kt[p]/Vt[p] ready; buffers p^1 no longer read
    if (wv == 0 && kb + 64 < T_)   // DMA next K tile during compute
      tdm_load_2d(ldsof(Kt[p ^ 1]), gKbase + (size_t)(kb + 64) * D_, D_, 64, D_, 4, 3);

    // S = Q @ K^T  (16 q x 64 keys)
    v8f sacc[4];
#pragma unroll
    for (int nt = 0; nt < 4; ++nt) sacc[nt] = ZERO8;
#pragma unroll
    for (int nt = 0; nt < 4; ++nt)
#pragma unroll
      for (int db = 0; db < 2; ++db) {
        v16bf kf = load_frag_b(Kt[p] + (nt * 16 + (lane & 15)) * FS, db * 16, lane);
        sacc[nt] = __builtin_amdgcn_wmma_f32_16x16x32_bf16(
            false, qfrag[db], false, kf, (short)0, sacc[nt], false, false);
      }

    // online softmax; row r lives across one 16-lane half in VGPR r
#pragma unroll
    for (int r = 0; r < 8; ++r) {
      float s0 = sacc[0][r], s1 = sacc[1][r], s2 = sacc[2][r], s3 = sacc[3][r];
      float mx = red16_max(fmaxf(fmaxf(s0, s1), fmaxf(s2, s3)));
      float mnew  = fmaxf(mst[r], mx);
      float alpha = __expf(mst[r] - mnew);
      float p0 = __expf(s0 - mnew);
      float p1 = __expf(s1 - mnew);
      float p2 = __expf(s2 - mnew);
      float p3 = __expf(s3 - mnew);
      lst[r] = lst[r] * alpha + red16_sum((p0 + p1) + (p2 + p3));
      mst[r] = mnew;
#pragma unroll
      for (int nt = 0; nt < 4; ++nt) oacc[nt][r] *= alpha;
      // store 4 contiguous bf16 at q-positions {c*4 .. c*4+3}, c = lane&15
      int prow = (lane >> 4) * 8 + r;
      u32x2 pk;
      pk.x = (unsigned int)f2bf(p0) | ((unsigned int)f2bf(p1) << 16);
      pk.y = (unsigned int)f2bf(p2) | ((unsigned int)f2bf(p3) << 16);
      *(u32x2*)(pbuf + prow * FS + (lane & 15) * 4) = pk;
    }
    // P tile is wave-private; same-wave DS ops are in-order -> wave-local drain
    asm volatile("s_wait_dscnt 0x0" ::: "memory");

    v16bf pf[2];
#pragma unroll
    for (int kc = 0; kc < 2; ++kc)
      pf[kc] = load_frag_a(pbuf + (lane & 15) * FS, kc * 16, lane);
#pragma unroll
    for (int nt = 0; nt < 4; ++nt)
#pragma unroll
      for (int kc = 0; kc < 2; ++kc) {
        v16bf vfr = load_frag_b(Vt[p] + (nt * 16 + (lane & 15)) * FS, kc * 16, lane);
        oacc[nt] = __builtin_amdgcn_wmma_f32_16x16x32_bf16(
            false, pf[kc], false, vfr, (short)0, oacc[nt], false, false);
      }
  }

  const int b = bh >> 4, h = bh & 15;
#pragma unroll
  for (int nt = 0; nt < 4; ++nt) {
    const int dcol = h * D_ + nt * 16 + (lane & 15);
#pragma unroll
    for (int r = 0; r < 8; ++r) {
      const int t = qw + (lane >> 4) * 8 + r;
      O[((size_t)b * T_ + t) * C_ + dcol] = f2bf(oacc[nt][r] / lst[r]);
    }
  }
}

// ---------------------------------------------------------------------------
// Launch: cvt(x, Wq, Wk, Wv, Wp) -> 3x GEMM (Q,K,V) -> flash -> GEMM(proj)
// Workspace layout (bytes), ~88 MiB total:
//   xbf 0 | Wq 16M | Wk 18M | Wv 20M | Wp 22M | Q 24M | K 40M | V 56M | Oat 72M
// ---------------------------------------------------------------------------
extern "C" void kernel_launch(void* const* d_in, const int* in_sizes, int n_in,
                              void* d_out, int out_size, void* d_ws, size_t ws_size,
                              hipStream_t stream) {
  const float* x  = (const float*)d_in[0];
  const float* Wk = (const float*)d_in[1];
  const float* Wq = (const float*)d_in[2];
  const float* Wv = (const float*)d_in[3];
  const float* Wp = (const float*)d_in[4];
  const float* bp = (const float*)d_in[5];

  char* w = (char*)d_ws;
  unsigned short* xbf = (unsigned short*)(w);
  unsigned short* wqb = (unsigned short*)(w + (16u << 20));
  unsigned short* wkb = (unsigned short*)(w + (18u << 20));
  unsigned short* wvb = (unsigned short*)(w + (20u << 20));
  unsigned short* wpb = (unsigned short*)(w + (22u << 20));
  unsigned short* Qb  = (unsigned short*)(w + (24u << 20));
  unsigned short* Kb  = (unsigned short*)(w + (40u << 20));
  unsigned short* Vb  = (unsigned short*)(w + (56u << 20));
  unsigned short* Ob  = (unsigned short*)(w + (72u << 20));

  attn_cvt_bf16<<<4096, 256, 0, stream>>>(x,  xbf, BT_ * C_);
  attn_cvt_bf16<<<512,  256, 0, stream>>>(Wq, wqb, C_ * C_);
  attn_cvt_bf16<<<512,  256, 0, stream>>>(Wk, wkb, C_ * C_);
  attn_cvt_bf16<<<512,  256, 0, stream>>>(Wv, wvb, C_ * C_);
  attn_cvt_bf16<<<512,  256, 0, stream>>>(Wp, wpb, C_ * C_);

  dim3 gg(C_ / 128, BT_ / 128);   // 8 x 64
  attn_gemm_wmma<0><<<gg, 256, 0, stream>>>(xbf, wqb, (void*)Qb, nullptr, 0.125f);
  attn_gemm_wmma<0><<<gg, 256, 0, stream>>>(xbf, wkb, (void*)Kb, nullptr, 1.0f);
  attn_gemm_wmma<0><<<gg, 256, 0, stream>>>(xbf, wvb, (void*)Vb, nullptr, 1.0f);

  dim3 fg(T_ / 128, B_ * H_);     // 16 x 64
  attn_flash_wmma<<<fg, 256, 0, stream>>>(Qb, Kb, Vb, Ob);

  attn_gemm_wmma<1><<<gg, 256, 0, stream>>>(Ob, wpb, d_out, bp, 1.0f);
}
